// GeneralGauntTensorProduct_42657615184401
// MI455X (gfx1250) — matrix operand order repaired and checked by reading, not compiled, not run on hardware
//
#include <hip/hip_runtime.h>
#include <hip/hip_bf16.h>

// ---------------- problem constants ----------------
#define RES_BETA  20
#define RES_ALPHA 19
#define NGRID     380            // RES_BETA * RES_ALPHA
#define GPAD      384            // padded grid (12 chunks of 32)
#define DX        25             // (LMAX_IN+1)^2
#define DXP       32             // padded K for bf16 WMMA
#define DOUT      81             // (2*LMAX_IN+1)^2
#define DOUTP     96             // padded out dim (6 tiles of 16)
#define NCH       128
#define MUL       128
#define BATCH     2048
#define PI_D      3.14159265358979323846

// ---------------- workspace layout (bytes) ----------------
#define OFF_YS   0u                                    // [GPAD][DXP]  bf16  (Yx == Yy)
#define OFF_YI   (OFF_YS + GPAD*DXP*2u)                // [DOUTP][GPAD] bf16 (Yinv^T)
#define OFF_WXT  (OFF_YI + (unsigned)(DOUTP*GPAD*2))   // [DX][NCH][MUL] bf16
#define OFF_WYT  (OFF_WXT + (unsigned)(DX*NCH*MUL*2))
#define OFF_XCW  (OFF_WYT + (unsigned)(DX*NCH*MUL*2))  // [BATCH*NCH][DXP] bf16
#define OFF_YCW  (OFF_XCW + (unsigned)(BATCH)*NCH*DXP*2u)

typedef __bf16 v16bf __attribute__((ext_vector_type(16)));
typedef __bf16 v8bf  __attribute__((ext_vector_type(8)));
typedef float  v8f   __attribute__((ext_vector_type(8)));

static __device__ __forceinline__ v8bf load8(const __bf16* p) {
  return *reinterpret_cast<const v8bf*>(p);
}
static __device__ __forceinline__ v16bf combine16(v8bf lo, v8bf hi) {
  return __builtin_shufflevector(lo, hi, 0,1,2,3,4,5,6,7,8,9,10,11,12,13,14,15);
}
static __device__ __forceinline__ v8f wmma_bf16(v16bf a, v16bf b, v8f c) {
#if defined(__HIP_DEVICE_COMPILE__)
  // D = A(16x32 bf16) * B(32x16 bf16) + C(16x16 f32)
  return __builtin_amdgcn_wmma_f32_16x16x32_bf16(false, a, false, b, (short)0, c, false, false);
#else
  (void)a; (void)b; return c;
#endif
}
static __device__ __forceinline__ int isqrt_i(int v) {
  int l = 0;
  while ((l + 1) * (l + 1) <= v) ++l;
  return l;
}

// ============================================================
// Kernel 0a: spherical-harmonic tables (device-computed constants).
// Ys  [GPAD][DXP] : B-operand for fx/fy GEMM,   Ys[g][d]  = Y_{L<=4}(g, d)
// Yit [DOUTP][GPAD]: B-operand for zc GEMM,     Yit[d][g] = Y_{L<=8}(g, d) * qw(g)
// ============================================================
__global__ void k_tables(__bf16* __restrict__ Ys, __bf16* __restrict__ Yit) {
  int g = threadIdx.x;               // launched with GPAD threads
  if (g >= GPAD) return;
  bool valid = (g < NGRID);
  int ib = g / RES_ALPHA;
  int ia = g - ib * RES_ALPHA;

  // Gauss-Legendre node/weight (Newton on P_20, fp64)
  double cb = 0.0, wgt = 0.0;
  if (valid) {
    const int n = RES_BETA;
    double xx = cos(PI_D * ((double)ib + 0.75) / ((double)n + 0.5));
    for (int it = 0; it < 64; ++it) {
      double p0 = 1.0, p1 = xx;
      for (int k = 2; k <= n; ++k) {
        double p2 = ((2.0 * k - 1.0) * xx * p1 - (k - 1.0) * p0) / (double)k;
        p0 = p1; p1 = p2;
      }
      double dp = (double)n * (xx * p1 - p0) / (xx * xx - 1.0);
      xx -= p1 / dp;
    }
    double p0 = 1.0, p1 = xx;
    for (int k = 2; k <= n; ++k) {
      double p2 = ((2.0 * k - 1.0) * xx * p1 - (k - 1.0) * p0) / (double)k;
      p0 = p1; p1 = p2;
    }
    double dp = (double)n * (xx * p1 - p0) / (xx * xx - 1.0);
    cb  = xx;
    wgt = 2.0 / ((1.0 - xx * xx) * dp * dp);
  }

  // associated Legendre up to l=8 (reference convention, Condon-Shortley)
  double P[9][9];
  for (int i = 0; i < 9; ++i)
    for (int j = 0; j < 9; ++j) P[i][j] = 0.0;
  double somx2 = sqrt(fmax(1.0 - cb * cb, 0.0));
  P[0][0] = 1.0;
  for (int m = 1; m <= 8; ++m) P[m][m] = -(2.0 * m - 1.0) * somx2 * P[m - 1][m - 1];
  for (int m = 0; m <= 8; ++m) {
    if (m + 1 <= 8) P[m + 1][m] = cb * (2.0 * m + 1.0) * P[m][m];
    for (int l = m + 2; l <= 8; ++l)
      P[l][m] = (cb * (2.0 * l - 1.0) * P[l - 1][m] - (l + m - 1.0) * P[l - 2][m]) / (double)(l - m);
  }
  double fact[17];
  fact[0] = 1.0;
  for (int i = 1; i < 17; ++i) fact[i] = fact[i - 1] * (double)i;

  double al = 2.0 * PI_D * (double)ia / (double)RES_ALPHA;
  double qw = wgt * (2.0 * PI_D / (double)RES_ALPHA);
  const double SQ2 = 1.4142135623730951;

  for (int d = 0; d < DXP; ++d) {
    float v = 0.0f;
    if (valid && d < DX) {
      int l = isqrt_i(d); int m = d - l * l - l; int am = m < 0 ? -m : m;
      double norm = sqrt((2.0 * l + 1.0) / (4.0 * PI_D) * fact[l - am] / fact[l + am]);
      double ang  = (m > 0) ? SQ2 * cos((double)m * al)
                   : ((m == 0) ? 1.0 : SQ2 * sin((double)am * al));
      v = (float)(norm * P[l][am] * ang);
    }
    Ys[g * DXP + d] = (__bf16)v;
  }
  for (int d = 0; d < DOUTP; ++d) {
    float v = 0.0f;
    if (valid && d < DOUT) {
      int l = isqrt_i(d); int m = d - l * l - l; int am = m < 0 ? -m : m;
      double norm = sqrt((2.0 * l + 1.0) / (4.0 * PI_D) * fact[l - am] / fact[l + am]);
      double ang  = (m > 0) ? SQ2 * cos((double)m * al)
                   : ((m == 0) ? 1.0 : SQ2 * sin((double)am * al));
      v = (float)(norm * P[l][am] * ang * qw);
    }
    Yit[d * GPAD + g] = (__bf16)v;
  }
}

// ============================================================
// Kernel 0b: pack weights:  Wt[d][c][u] = W[lix[d]][u][c] / sqrt(MUL)  (bf16)
// ============================================================
__global__ void k_weights(const float* __restrict__ Wx, const float* __restrict__ Wy,
                          __bf16* __restrict__ Wxt, __bf16* __restrict__ Wyt) {
  const int total = DX * NCH * MUL;
  int idx = blockIdx.x * blockDim.x + threadIdx.x;
  if (idx >= 2 * total) return;
  const float scale = 0.08838834764831845f; // 1/sqrt(128)
  int t = idx;
  const float* src = Wx; __bf16* dst = Wxt;
  if (t >= total) { t -= total; src = Wy; dst = Wyt; }
  int d   = t / (NCH * MUL);
  int rem = t - d * (NCH * MUL);
  int c   = rem / MUL;
  int u   = rem - c * MUL;
  int l   = isqrt_i(d);
  dst[t] = (__bf16)(src[(size_t)(l * MUL + u) * NCH + c] * scale);
}

// ============================================================
// Kernel 1: xc[b,c,d] = sum_u x[b,u,d] * Wt[d][c][u]  (per-d GEMM, K=128)
// grid (BATCH/16, DX, 2); 8 waves; wave w -> c-tile w*16
// Output: bf16, padded [row][DXP] for direct A-operand loads in kernel 2.
// ============================================================
__global__ void __launch_bounds__(256) k_proj(const float* __restrict__ x,
                                              const float* __restrict__ y,
                                              const __bf16* __restrict__ Wxt,
                                              const __bf16* __restrict__ Wyt,
                                              __bf16* __restrict__ xcw,
                                              __bf16* __restrict__ ycw) {
  __shared__ __align__(16) __bf16 Xa[16 * MUL];   // 4 KB A tile, shared by all waves
  int b0 = blockIdx.x * 16;
  int d  = blockIdx.y;
  int which = blockIdx.z;
  const float*  src = which ? y : x;
  const __bf16* W   = which ? Wyt : Wxt;
  __bf16*       dst = which ? ycw : xcw;

  // stage x[b0..b0+16, :, d] -> LDS bf16 (strided gather; x is L2-resident)
  for (int e = threadIdx.x; e < 16 * MUL; e += 256) {
    int m = e >> 7, u = e & (MUL - 1);
    Xa[m * MUL + u] = (__bf16)src[((size_t)(b0 + m) * MUL + u) * DX + d];
  }
  __syncthreads();

  int lane = threadIdx.x & 31;
  int wave = threadIdx.x >> 5;
  int c0   = wave * 16;
  int hi   = lane >> 4;
  int ln   = lane & 15;
  int ak   = hi ? 8 : 0;    // A: per-lane K base
  int bk   = hi ? 8 : 0;    // B: per-lane K base

  const __bf16* abase = Xa + ln * MUL + ak;
  const __bf16* wbase = W + (size_t)d * NCH * MUL + (size_t)(c0 + ln) * MUL + bk;

  v8f acc = {};
#pragma unroll
  for (int kc = 0; kc < MUL; kc += 32) {
    v16bf A = combine16(load8(abase + kc), load8(abase + kc + 16));
    v16bf B = combine16(load8(wbase + kc), load8(wbase + kc + 16));
    acc = wmma_bf16(A, B, acc);
  }

#pragma unroll
  for (int r = 0; r < 8; ++r) {
    int m = ak + r;
    size_t row = (size_t)(b0 + m) * NCH + (c0 + ln);
    dst[row * DXP + d] = (__bf16)acc[r];
    if (d == 0) {                          // zero the K padding once per row
#pragma unroll
      for (int dp = DX; dp < DXP; ++dp) dst[row * DXP + dp] = (__bf16)0.0f;
    }
  }
}

// ============================================================
// Kernel 2: fused  fx = xc*Ys^T, fy = yc*Ys^T, fz = fx.*fy, zc = fz*Yinv
// One 16-row (b,c) tile per wave; fz kept on-WGP via bf16 LDS relayout.
// 120 WMMAs / wave; zc (16x96) accumulated in registers over all 380 g.
// ============================================================
__global__ void __launch_bounds__(256) k_fused(const __bf16* __restrict__ xcw,
                                               const __bf16* __restrict__ ycw,
                                               const __bf16* __restrict__ Ys,
                                               const __bf16* __restrict__ Yit,
                                               float* __restrict__ out) {
  __shared__ __align__(16) __bf16 fzbuf[8][16 * DXP];  // 1 KB per wave, private
  int lane = threadIdx.x & 31;
  int wave = threadIdx.x >> 5;
  int row0 = (blockIdx.x * 8 + wave) * 16;
  int hi = lane >> 4;
  int ln = lane & 15;
  int ak = hi ? 8 : 0;
  int bk = hi ? 8 : 0;

  const __bf16* axp = xcw + (size_t)(row0 + ln) * DXP + ak;
  const __bf16* ayp = ycw + (size_t)(row0 + ln) * DXP + ak;
  v16bf Ax = combine16(load8(axp), load8(axp + 16));
  v16bf Ay = combine16(load8(ayp), load8(ayp + 16));

  v8f acc[6];
#pragma unroll
  for (int nt = 0; nt < 6; ++nt) acc[nt] = (v8f){};

  __bf16* fzw = &fzbuf[wave][0];

  for (int gc = 0; gc < GPAD; gc += 32) {
#pragma unroll
    for (int sub = 0; sub < 2; ++sub) {
      int g = gc + sub * 16 + ln;
      const __bf16* bxp = Ys + (size_t)g * DXP + bk;
      v16bf B = combine16(load8(bxp), load8(bxp + 16));   // Yx == Yy: shared B
      v8f zero = {};
      v8f fx = wmma_bf16(Ax, B, zero);
      v8f fy = wmma_bf16(Ay, B, zero);
#pragma unroll
      for (int r = 0; r < 8; ++r) {
        // D-layout (m = ak+r across VGPRs, n = ln across lanes) -> A-layout [m][k]
        fzw[(ak + r) * DXP + sub * 16 + ln] = (__bf16)(fx[r] * fy[r]);
      }
    }
    // reload fz as A operand (in-wave DS ordering; no barrier needed)
    const __bf16* fzp = fzw + ln * DXP + ak;
    v16bf Afz = combine16(load8(fzp), load8(fzp + 16));
#pragma unroll
    for (int nt = 0; nt < 6; ++nt) {
      const __bf16* bip = Yit + (size_t)(nt * 16 + ln) * GPAD + gc + bk;
      v16bf Bi = combine16(load8(bip), load8(bip + 16));
      acc[nt] = wmma_bf16(Afz, Bi, acc[nt]);
    }
  }

  // epilogue: out[row][d], d < 81
#pragma unroll
  for (int nt = 0; nt < 6; ++nt) {
    int d = nt * 16 + ln;
#pragma unroll
    for (int r = 0; r < 8; ++r) {
      int m = ak + r;
      if (d < DOUT) out[(size_t)(row0 + m) * DOUT + d] = acc[nt][r];
    }
  }
}

// ============================================================
extern "C" void kernel_launch(void* const* d_in, const int* in_sizes, int n_in,
                              void* d_out, int out_size, void* d_ws, size_t ws_size,
                              hipStream_t stream) {
  (void)in_sizes; (void)n_in; (void)out_size; (void)ws_size;
  const float* x  = (const float*)d_in[0];
  const float* y  = (const float*)d_in[1];
  const float* Wx = (const float*)d_in[2];
  const float* Wy = (const float*)d_in[3];
  float* out = (float*)d_out;

  char* ws = (char*)d_ws;
  __bf16* Ys  = (__bf16*)(ws + OFF_YS);
  __bf16* Yit = (__bf16*)(ws + OFF_YI);
  __bf16* Wxt = (__bf16*)(ws + OFF_WXT);
  __bf16* Wyt = (__bf16*)(ws + OFF_WYT);
  __bf16* xcw = (__bf16*)(ws + OFF_XCW);
  __bf16* ycw = (__bf16*)(ws + OFF_YCW);

  k_tables<<<1, GPAD, 0, stream>>>(Ys, Yit);
  k_weights<<<(2 * DX * NCH * MUL + 255) / 256, 256, 0, stream>>>(Wx, Wy, Wxt, Wyt);
  k_proj<<<dim3(BATCH / 16, DX, 2), 256, 0, stream>>>(x, y, Wxt, Wyt, xcw, ycw);
  k_fused<<<(BATCH * NCH) / (8 * 16), 256, 0, stream>>>(xcw, ycw, Ys, Yit, out);
}
